// HandwritingRNN_22600117911807
// MI455X (gfx1250) — compile-verified, matrix operand order checked by reading, not run on hardware
//
#include <hip/hip_runtime.h>

// ---------------- problem constants ----------------
#define B_   128
#define T_   600
#define U_   50
#define H_   400
#define V_   80
#define K_   10
#define OUT_ 121
#define KTOT 512        // padded A width: [x(3) | w(80) | pad..96 | h(400) | pad..512]
#define HOFF 96         // h starts at K index 96
#define HSTR 416        // padded hidden stride of hs panel (bf16), pad cols 400..415 are zero
#define NWG  25         // workgroups in recurrent kernel (16 hidden units each)
#define NCOL 64         // z columns per WG: 4 gates x 16 hidden units

// LDS layout byte offsets (recurrent kernel)
#define LDS_HBUF_OFF ((NCOL*KTOT + 32*HSTR + B_*96)*2 + (B_*32 + B_*K_ + B_*U_ + B_*V_)*4) // 204800
#define SMEM_RNN_SZ  (LDS_HBUF_OFF + B_*HSTR*2)                                            // 311296
#define SMEM_OUT_SZ  (128*HSTR*2)                                                          // 106496

typedef __attribute__((ext_vector_type(16))) __bf16 v16bf;
typedef __attribute__((ext_vector_type(8)))  float  v8f;
typedef unsigned int u32x4 __attribute__((ext_vector_type(4)));
typedef int          i32x4 __attribute__((ext_vector_type(4)));
typedef int          i32x8 __attribute__((ext_vector_type(8)));

union Frag { v16bf v; uint4 q[2]; unsigned short s[16]; };

__device__ __forceinline__ unsigned short f2bf(float f) {
  unsigned u = __float_as_uint(f);
  u += 0x7fffu + ((u >> 16) & 1u);        // round-to-nearest-even
  return (unsigned short)(u >> 16);
}
__device__ __forceinline__ float sigm(float z) { return 1.f / (1.f + __expf(-z)); }

// Load a 16x32 bf16 A-fragment (or matching B-fragment from a column-major
// panel) for one lane, per the CDNA5 WMMA VGPR layout:
//   lanes 0-15 : K = kbase+[0..7] and kbase+16+[0..7]
//   lanes 16-31: K = kbase+8+[0..7] and kbase+24+[0..7]
__device__ __forceinline__ v16bf load_frag(const unsigned short* p, int kbase, int koff) {
  Frag f;
  f.q[0] = *(const uint4*)(p + kbase + koff);
  f.q[1] = *(const uint4*)(p + kbase + 16 + koff);
  return f.v;
}

// ---- Tensor Data Mover: DMA a 2-D bf16 tile [rows x HSTR] global -> LDS.
// D# per CDNA5 ISA §8.3/8.4: group0 {count=1, lds_addr, global_addr, type=2},
// group1 {data_size=2B, tensor_dim0=HSTR, tensor_dim1=rows, tile_dim0=HSTR,
// tile_dim1=rows, tensor_dim0_stride=HSTR}; groups 2/3 zero (<=2-D tensor).
__device__ __forceinline__ void tdm_load_panel(const unsigned short* gsrc, unsigned lds_off) {
  unsigned long long ga = (unsigned long long)(uintptr_t)gsrc;
  u32x4 g0;
  g0[0] = 1u;                                                   // count=1, user D#
  g0[1] = lds_off;                                              // lds_addr (bytes)
  g0[2] = (unsigned)(ga & 0xffffffffu);                         // global_addr[31:0]
  g0[3] = (unsigned)((ga >> 32) & 0x01ffffffu) | (2u << 30);    // addr[56:32] | type=2
  i32x8 g1;
  g1[0] = (1 << 16);                                            // data_size=1 (2 bytes), no multicast
  g1[1] = (int)((HSTR & 0xffff) << 16);                         // tensor_dim0[15:0] (lo bits)
  g1[2] = (int)((B_ & 0xffff) << 16);                           // tensor_dim0 hi=0 | tensor_dim1 lo
  g1[3] = (int)((HSTR & 0xffff) << 16);                         // tensor_dim1 hi=0 | tile_dim0
  g1[4] = B_;                                                   // tile_dim1 | tile_dim2=0
  g1[5] = HSTR;                                                 // tensor_dim0_stride[31:0]
  g1[6] = 0;                                                    // stride hi / dim1_stride lo
  g1[7] = 0;                                                    // dim1_stride hi (unused, 2-D)
  i32x4 z4; z4[0] = z4[1] = z4[2] = z4[3] = 0;
#if __has_include(<hip/amd_detail/amd_gfx1250_TDM.h>)
  // amdgpu-toolchain (clang-23 / therock-10.0): 6-arg builtin
  i32x8 z8;
  #pragma unroll
  for (int i = 0; i < 8; ++i) z8[i] = 0;
  __builtin_amdgcn_tensor_load_to_lds(g0, g1, z4, z4, z8, 0);
#else
  // ROCm 7.2 (clang-22): 5-arg builtin
  __builtin_amdgcn_tensor_load_to_lds(g0, g1, z4, z4, 0);
#endif
}

// ---------------- init: zero hs panel (slot 0 = h_{-1} = 0, pad cols = 0), reset barrier
__global__ void k_init(unsigned short* hs, int* bar, size_t n) {
  size_t i  = (size_t)blockIdx.x * blockDim.x + threadIdx.x;
  size_t st = (size_t)gridDim.x * blockDim.x;
  for (; i < n; i += st) hs[i] = 0;
  if (blockIdx.x == 0 && threadIdx.x == 0) *bar = 0;
}

// ---------------- pack W_ih/W_hh into one padded bf16 matrix Wz[KTOT][1600]
__global__ void k_pack(const float* __restrict__ W_ih, const float* __restrict__ W_hh,
                       unsigned short* __restrict__ Wz) {
  int i = blockIdx.x * 256 + threadIdx.x;          // grid sized exactly KTOT*1600/256
  if (i >= KTOT * 1600) return;
  int k = i / 1600, n = i - k * 1600;
  float v = 0.f;
  if (k < 83)                          v = W_ih[k * 1600 + n];
  else if (k >= HOFF && k < HOFF + H_) v = W_hh[(k - HOFF) * 1600 + n];
  Wz[i] = f2bf(v);
}

// ---------------- persistent recurrent kernel: 25 WGs, global spin barrier per step
__global__ __launch_bounds__(256, 1)
void k_rnn(const float* __restrict__ x, const int* __restrict__ c,
           const int* __restrict__ c_len,
           const float* __restrict__ b_lstm, const float* __restrict__ W_attn,
           const float* __restrict__ b_attn,
           const unsigned short* __restrict__ Wz,
           unsigned short* __restrict__ hs, int* bar)
{
  extern __shared__ unsigned char smem[];
  unsigned short* Wl  = (unsigned short*)smem;        // [64][512]  z-weight slice, col-major
  unsigned short* Wa  = Wl + NCOL * KTOT;             // [32][416]  attn weights, col-major
  unsigned short* axw = Wa + 32 * HSTR;               // [128][96]  bf16 [x_t | w | pad] panel
  float* apb  = (float*)(axw + B_ * 96);              // [128][32]  attn pre-acts / alpha,beta
  float* kap  = apb  + B_ * 32;                       // [128][10]  kappa state
  float* phib = kap  + B_ * K_;                       // [128][50]  phi
  float* wbuf = phib + B_ * U_;                       // [128][80]  window accum
  unsigned short* hbuf = (unsigned short*)(smem + LDS_HBUF_OFF); // [128][416] TDM-staged h

  const int tid  = threadIdx.x;
  const int j0   = blockIdx.x * 16;                   // first hidden unit owned
  const int lane = tid & 31;
  const int wv   = tid >> 5;                          // wave id == M-tile (batch rows 16*wv..)
  const int mrow = (wv << 4) + (lane & 15);           // A-matrix row this lane feeds
  const int koff = (lane & 16) ? 8 : 0;
  const int nc   = lane & 15;                         // B-matrix column within an N-tile

  // stage z-weight slice: gather columns {gate*400 + j0 + j}, transpose to col-major
  for (int i = tid; i < NCOL * KTOT; i += 256) {
    int n = i >> 9, k = i & (KTOT - 1);
    int gate = n >> 4, j = n & 15;
    Wl[i] = Wz[k * 1600 + gate * H_ + j0 + j];
  }
  // stage attention weights (f32 -> bf16, zero-padded to 32 x 416), col-major
  for (int i = tid; i < 32 * HSTR; i += 256) {
    int n = i / HSTR, k = i - n * HSTR;
    Wa[i] = f2bf((n < 30 && k < H_) ? W_attn[k * 30 + n] : 0.f);
  }
  for (int i = tid; i < B_ * 96; i += 256) axw[i] = 0;    // w(t=0)=0, pads=0
  for (int i = tid; i < B_ * K_; i += 256) kap[i] = 0.f;  // kappa(0)=0
  for (int i = tid; i < B_ * HSTR; i += 256) hbuf[i] = 0; // h_{-1}=0
  __syncthreads();

  v8f csr;                                            // persistent 16x16 cell-state block
  #pragma unroll
  for (int r = 0; r < 8; ++r) csr[r] = 0.f;

  const float bi = b_lstm[0 * H_ + j0 + nc];
  const float bf = b_lstm[1 * H_ + j0 + nc];
  const float bg = b_lstm[2 * H_ + j0 + nc];
  const float bo = b_lstm[3 * H_ + j0 + nc];
  const float ba0 = (nc      < 30) ? b_attn[nc]      : 0.f;
  const float ba1 = (nc + 16 < 30) ? b_attn[nc + 16] : 0.f;

  for (int t = 0; t < T_; ++t) {
    // ---- phase 0: stream x_t into the LDS A-panel
    if (tid < B_) {
      const float* xp = x + ((size_t)tid * T_ + t) * 3;
      axw[tid * 96 + 0] = f2bf(xp[0]);
      axw[tid * 96 + 1] = f2bf(xp[1]);
      axw[tid * 96 + 2] = f2bf(xp[2]);
      if (t + 1 < T_) __builtin_prefetch(xp + 3, 0, 0);   // global_prefetch next x_t
    }
    __syncthreads();

    // ---- phase 1: z = [x,w,h_prev] @ Wslice  (16 k-steps x 4 gate tiles of bf16 WMMA)
    //      h_prev comes from the TDM-staged LDS panel (loaded last iteration).
    v8f a_i, a_f, a_g, a_o;
    #pragma unroll
    for (int r = 0; r < 8; ++r) { a_i[r] = 0; a_f[r] = 0; a_g[r] = 0; a_o[r] = 0; }
    const unsigned short* hrow = hbuf + mrow * HSTR;
    const unsigned short* arow = axw + mrow * 96;
    #pragma unroll
    for (int ks = 0; ks < 16; ++ks) {
      v16bf a = (ks < 3) ? load_frag(arow, ks * 32,        koff)
                         : load_frag(hrow, ks * 32 - HOFF, koff);
      v16bf b0 = load_frag(Wl + (nc     ) * KTOT, ks * 32, koff);
      v16bf b1 = load_frag(Wl + (nc + 16) * KTOT, ks * 32, koff);
      v16bf b2 = load_frag(Wl + (nc + 32) * KTOT, ks * 32, koff);
      v16bf b3 = load_frag(Wl + (nc + 48) * KTOT, ks * 32, koff);
      a_i = __builtin_amdgcn_wmma_f32_16x16x32_bf16(false, a, false, b0, (short)0, a_i, false, false);
      a_f = __builtin_amdgcn_wmma_f32_16x16x32_bf16(false, a, false, b1, (short)0, a_f, false, false);
      a_g = __builtin_amdgcn_wmma_f32_16x16x32_bf16(false, a, false, b2, (short)0, a_g, false, false);
      a_o = __builtin_amdgcn_wmma_f32_16x16x32_bf16(false, a, false, b3, (short)0, a_o, false, false);
    }

    // ---- phase 2: LSTM update (cell state lives in registers), publish h bf16
    unsigned short* hn = hs + ((size_t)(t + 1) * B_) * HSTR;
    #pragma unroll
    for (int r = 0; r < 8; ++r) {
      int bb = (wv << 4) + r + ((lane >> 4) << 3);       // C/D row mapping
      float cs = sigm(a_f[r] + bf) * csr[r] + sigm(a_i[r] + bi) * tanhf(a_g[r] + bg);
      csr[r] = cs;
      float h = sigm(a_o[r] + bo) * tanhf(cs);
      hn[(size_t)bb * HSTR + j0 + nc] = f2bf(h);
    }
    __threadfence();
    __syncthreads();

    // ---- phase 3: device-wide step barrier, then TDM-stage the new h panel
    //      (all waves are parked between barriers, so hbuf is safe to overwrite)
    if (tid == 0) {
      atomicAdd(bar, 1);
      const int tgt = NWG * (t + 1);
      while (__hip_atomic_load(bar, __ATOMIC_ACQUIRE, __HIP_MEMORY_SCOPE_AGENT) < tgt)
        __builtin_amdgcn_s_sleep(1);
      tdm_load_panel(hs + ((size_t)(t + 1) * B_) * HSTR, (unsigned)LDS_HBUF_OFF);
      __builtin_amdgcn_s_wait_tensorcnt((short)0);
    }
    __syncthreads();

    // ---- phase 4: attention pre-acts ap = h @ W_attn (WMMA from LDS panel)
    v8f p0, p1;
    #pragma unroll
    for (int r = 0; r < 8; ++r) { p0[r] = 0; p1[r] = 0; }
    #pragma unroll
    for (int ks = 0; ks < 13; ++ks) {
      v16bf a  = load_frag(hrow, ks * 32, koff);
      v16bf b0 = load_frag(Wa + (nc     ) * HSTR, ks * 32, koff);
      v16bf b1 = load_frag(Wa + (nc + 16) * HSTR, ks * 32, koff);
      p0 = __builtin_amdgcn_wmma_f32_16x16x32_bf16(false, a, false, b0, (short)0, p0, false, false);
      p1 = __builtin_amdgcn_wmma_f32_16x16x32_bf16(false, a, false, b1, (short)0, p1, false, false);
    }
    #pragma unroll
    for (int r = 0; r < 8; ++r) {
      int bb = (wv << 4) + r + ((lane >> 4) << 3);
      apb[bb * 32 + nc]      = p0[r] + ba0;
      apb[bb * 32 + nc + 16] = p1[r] + ba1;
    }
    __syncthreads();

    // ---- phase 5: alpha/beta/kappa (one thread per (b,k) mixture)
    for (int p = tid; p < B_ * K_; p += 256) {
      int bb = p / K_, k = p - bb * K_;
      float ah = apb[bb * 32 + k];
      float bh = apb[bb * 32 + 10 + k];
      float kh = apb[bb * 32 + 20 + k];
      kap[p] += __expf(kh);
      apb[bb * 32 + k]      = __expf(ah);                // alpha
      apb[bb * 32 + 10 + k] = __expf(bh);                // beta
    }
    __syncthreads();

    // ---- phase 6: phi[b][u] = sum_k alpha * exp(-beta * (kappa-u)^2), masked
    for (int p = tid; p < B_ * U_; p += 256) {
      int bb = p / U_, u = p - bb * U_;
      float s = 0.f;
      if (u < c_len[bb]) {
        #pragma unroll
        for (int k = 0; k < K_; ++k) {
          float d = kap[bb * K_ + k] - (float)u;
          s += apb[bb * 32 + k] * __expf(-apb[bb * 32 + 10 + k] * d * d);
        }
      }
      phib[p] = s;
    }
    __syncthreads();

    // ---- phase 7: w[b][v] = sum_u phi * onehot(c)  (one thread per batch row)
    if (tid < B_) {
      float* wrow = wbuf + tid * V_;
      for (int v = 0; v < V_; ++v) wrow[v] = 0.f;
      const int* crow = c + tid * U_;
      for (int u = 0; u < U_; ++u) wrow[crow[u]] += phib[tid * U_ + u];
      unsigned short* aw = axw + tid * 96 + 3;
      for (int v = 0; v < V_; ++v) aw[v] = f2bf(wrow[v]);
    }
    __syncthreads();
  }
}

// ---------------- epilogue: out = hs @ W_out + b_out  (76800 x 416 x 128-padded)
__global__ __launch_bounds__(256, 1)
void k_out(const unsigned short* __restrict__ hs, const float* __restrict__ W_out,
           const float* __restrict__ b_out, float* __restrict__ out)
{
  extern __shared__ unsigned char smem[];
  unsigned short* Wol = (unsigned short*)smem;           // [128][416] col-major bf16
  const int tid = threadIdx.x, lane = tid & 31, wv = tid >> 5;
  for (int i = tid; i < 128 * HSTR; i += 256) {
    int n = i / HSTR, k = i - n * HSTR;
    Wol[i] = f2bf((n < OUT_ && k < H_) ? W_out[k * OUT_ + n] : 0.f);
  }
  __syncthreads();

  const int mtile = blockIdx.x;                          // 4800 row tiles of (t,b)
  const int gm = (mtile << 4) + (lane & 15);
  const int tt = gm >> 7, bb = gm & 127;                 // row = t*128 + b
  const unsigned short* arow = hs + ((size_t)(tt + 1) * B_ + bb) * HSTR;
  const int koff = (lane & 16) ? 8 : 0;
  const int nc = (wv << 4) + (lane & 15);                // N-tile = wave id

  v8f acc;
  #pragma unroll
  for (int r = 0; r < 8; ++r) acc[r] = 0;
  #pragma unroll
  for (int ks = 0; ks < 13; ++ks) {
    v16bf a = load_frag(arow, ks * 32, koff);
    v16bf b = load_frag(Wol + nc * HSTR, ks * 32, koff);
    acc = __builtin_amdgcn_wmma_f32_16x16x32_bf16(false, a, false, b, (short)0, acc, false, false);
  }
  if (nc < OUT_) {
    const float bo = b_out[nc];
    #pragma unroll
    for (int r = 0; r < 8; ++r) {
      int g = (mtile << 4) + r + ((lane >> 4) << 3);
      int t2 = g >> 7, b2 = g & 127;
      out[((size_t)b2 * T_ + t2) * OUT_ + nc] = acc[r] + bo;   // reference order [B,T,121]
    }
  }
}

// ---------------- host ----------------
extern "C" void kernel_launch(void* const* d_in, const int* in_sizes, int n_in,
                              void* d_out, int out_size, void* d_ws, size_t ws_size,
                              hipStream_t stream) {
  (void)in_sizes; (void)n_in; (void)out_size; (void)ws_size;
  const float* x      = (const float*)d_in[0];
  const int*   c      = (const int*)  d_in[1];
  /* d_in[2] = x_len: unused by the reference */
  const int*   c_len  = (const int*)  d_in[3];
  const float* W_ih   = (const float*)d_in[4];
  const float* W_hh   = (const float*)d_in[5];
  const float* b_lstm = (const float*)d_in[6];
  const float* W_attn = (const float*)d_in[7];
  const float* b_attn = (const float*)d_in[8];
  const float* W_out  = (const float*)d_in[9];
  const float* b_out  = (const float*)d_in[10];
  float* out = (float*)d_out;

  // workspace layout: hs panel (bf16, T+1 slots, padded rows) | packed Wz bf16 | barrier
  unsigned short* hs = (unsigned short*)d_ws;
  const size_t hs_elems = (size_t)(T_ + 1) * B_ * HSTR;
  unsigned short* Wz = hs + hs_elems;
  int* bar = (int*)(Wz + (size_t)KTOT * 1600);

  hipFuncSetAttribute(reinterpret_cast<const void*>(k_rnn),
                      hipFuncAttributeMaxDynamicSharedMemorySize, SMEM_RNN_SZ);
  hipFuncSetAttribute(reinterpret_cast<const void*>(k_out),
                      hipFuncAttributeMaxDynamicSharedMemorySize, SMEM_OUT_SZ);

  k_init<<<2048, 256, 0, stream>>>(hs, bar, hs_elems);
  k_pack<<<(KTOT * 1600) / 256, 256, 0, stream>>>(W_ih, W_hh, Wz);
  k_rnn <<<NWG, 256, SMEM_RNN_SZ, stream>>>(x, c, c_len, b_lstm, W_attn, b_attn, Wz, hs, bar);
  k_out <<<(T_ * B_) / 16, 256, SMEM_OUT_SZ, stream>>>(hs, W_out, b_out, out);
}
